// SE3Conv_10282151706923
// MI455X (gfx1250) — matrix operand order repaired and checked by reading, not compiled, not run on hardware
//
#include <hip/hip_runtime.h>

typedef __attribute__((ext_vector_type(16))) _Float16 v16h;
typedef __attribute__((ext_vector_type(4)))  _Float16 v4h;
typedef __attribute__((ext_vector_type(8)))  float    v8f;
typedef __attribute__((ext_vector_type(4)))  float    v4f;

// ---------------------------------------------------------------------------
// Stage 1: build kr/ki (64, 32, 27) f32 from the per-(l,l1) complex matmuls.
// One thread per pre-reshape element; remap via C-order flat index math.
// ---------------------------------------------------------------------------
__global__ void se3_build_kernel(
    const float* __restrict__ w00r, const float* __restrict__ w00i,
    const float* __restrict__ w01r, const float* __restrict__ w01i,
    const float* __restrict__ w10r, const float* __restrict__ w10i,
    const float* __restrict__ w11r, const float* __restrict__ w11i,
    const float* __restrict__ f00r, const float* __restrict__ f00i,
    const float* __restrict__ f10r, const float* __restrict__ f10i,
    const float* __restrict__ f01r, const float* __restrict__ f01i,
    const float* __restrict__ f11r, const float* __restrict__ f11i,
    float* __restrict__ kr, float* __restrict__ ki)
{
    int idx = blockIdx.x * blockDim.x + threadIdx.x;
    if (idx >= 55296) return;

    int l, l1, base;
    const float *wr, *wi, *fr, *fi;
    if (idx < 3456)       { l = 0; l1 = 0; base = 0;     wr = w00r; wi = w00i; fr = f00r; fi = f00i; }
    else if (idx < 13824) { l = 0; l1 = 1; base = 3456;  wr = w01r; wi = w01i; fr = f10r; fi = f10i; }
    else if (idx < 24192) { l = 1; l1 = 0; base = 13824; wr = w10r; wi = w10i; fr = f01r; fi = f01i; }
    else                  { l = 1; l1 = 1; base = 24192; wr = w11r; wi = w11i; fr = f11r; fi = f11i; }

    const int kdim = (2 * l + 1) * (2 * l1 + 1) * 27;
    int e   = idx - base;
    int o16 = e / (8 * kdim);
    int rem = e % (8 * kdim);
    int c8  = rem / kdim;
    int j   = rem % kdim;

    float sr = 0.0f, si = 0.0f;
    #pragma unroll
    for (int r = 0; r < 4; r++) {
        float a = wr[(o16 * 8 + c8) * 4 + r];
        float b = wi[(o16 * 8 + c8) * 4 + r];
        float c = fr[r * kdim + j];
        float d = fi[r * kdim + j];
        sr += a * c - b * d;
        si += a * d + b * c;
    }

    // C-order reshape (16,8,kdim) -> (Ob, Cb, 27), then place in global (64,32,27)
    int Cb    = (2 * l1 + 1) * 8;
    int flat  = (o16 * 8 + c8) * kdim + j;
    int O_idx = flat / (Cb * 27);
    int r2    = flat % (Cb * 27);
    int C_idx = r2 / 27;
    int tap   = r2 % 27;
    int O = (l ? 16 : 0) + O_idx;
    int C = (l1 ? 8 : 0) + C_idx;
    kr[(O * 32 + C) * 27 + tap] = sr;
    ki[(O * 32 + C) * 27 + tap] = si;
}

// ---------------------------------------------------------------------------
// Stage 2: pack combined complex weights into f16 WMMA B-fragments.
// Combined GEMM: N=128 (n<64 -> yr, n>=64 -> yi), K=64 (k<32 -> xr, k>=32 -> xi).
//   yr weights: [ kr | -ki ],  yi weights: [ ki | kr ].
// B layout (32x16 f16, wave32): lane%16 = N column, 16 halves = K rows (table order, 8*(lane/16) offset).
// wpack[(((tap*2 + kc)*8 + nt)*32 + lane)*16 + j]
// ---------------------------------------------------------------------------
__global__ void se3_pack_weights(const float* __restrict__ kr,
                                 const float* __restrict__ ki,
                                 _Float16* __restrict__ wpack)
{
    int tid = blockIdx.x * blockDim.x + threadIdx.x;  // 27*2*8*32 = 13824
    if (tid >= 13824) return;
    int lane = tid & 31;
    int nt   = (tid >> 5) & 7;
    int kc   = (tid >> 8) & 1;
    int tap  = tid >> 9;
    int n    = nt * 16 + (lane & 15);
    int kb   = kc * 32 + (lane >> 4) * 8;

    _Float16* dst = wpack + (size_t)tid * 16;
    #pragma unroll
    for (int j = 0; j < 16; j++) {
        int k    = kb + (j & 7) + (j >> 3) * 16;
        int c    = k & 31;
        int part = n >> 6;      // 0 = yr, 1 = yi
        int o    = n & 63;
        float vr = kr[(o * 32 + c) * 27 + tap];
        float vi = ki[(o * 32 + c) * 27 + tap];
        float v  = (part == 0) ? ((k < 32) ? vr : -vi)
                               : ((k < 32) ? vi :  vr);
        dst[j] = (_Float16)v;
    }
}

// ---------------------------------------------------------------------------
// Stage 2b: one-shot activation conversion f32 -> f16 (same channel-last
// layout). Halves the per-tap A-matrix traffic in the conv loop and removes
// 64 VALU conversion ops per tap from the WMMA inner loop.
// ---------------------------------------------------------------------------
__global__ void se3_cvt_x_f16(const float* __restrict__ xr,
                              const float* __restrict__ xi,
                              _Float16* __restrict__ xrh,
                              _Float16* __restrict__ xih)
{
    int t = blockIdx.x * blockDim.x + threadIdx.x;  // 1048576/4 = 262144 threads
    v4f a = ((const v4f*)xr)[t];
    v4f b = ((const v4f*)xi)[t];
    v4h ha, hb;
    #pragma unroll
    for (int q = 0; q < 4; q++) { ha[q] = (_Float16)a[q]; hb[q] = (_Float16)b[q]; }
    ((v4h*)xrh)[t] = ha;
    ((v4h*)xih)[t] = hb;
}

// ---------------------------------------------------------------------------
// Stage 3: implicit-GEMM complex conv3d via v_wmma_f32_16x16x32_f16.
// Wave: 16 spatial rows x 128 outputs (8 accumulator tiles).
// Workgroup: 8 waves; per-tap B slice (16 KB) double-buffered through LDS.
// ---------------------------------------------------------------------------
__global__ __launch_bounds__(256)
void se3_conv_wmma(const _Float16* __restrict__ xrh, const _Float16* __restrict__ xih,
                   const _Float16* __restrict__ wpack, float* __restrict__ out)
{
    __shared__ __align__(32) _Float16 lds[2][8192];  // 2 x 16 KB tap slices

    const int tid      = threadIdx.x;
    const int lane     = tid & 31;
    const int wave     = tid >> 5;
    const int tileBase = (blockIdx.x * 8 + wave) * 16;
    const int laneM    = lane & 15;
    const int hiHalf   = lane >> 4;           // 0: K 0..7/16..23, 1: K 8..15/24..31
    const int s        = tileBase + laneM;    // this lane's spatial row (M)
    const int w0 = s & 31, h0 = (s >> 5) & 31, d0 = s >> 10;
    const int kOff = hiHalf * 8;

    v8f acc[8];
    #pragma unroll
    for (int i = 0; i < 8; i++) acc[i] = v8f{};

    // stage tap 0 into buffer 0 (each thread copies 64 B)
    {
        const uint4* src = (const uint4*)wpack;
        uint4* dst = (uint4*)(&lds[0][0]);
        #pragma unroll
        for (int i = 0; i < 4; i++) dst[tid * 4 + i] = src[tid * 4 + i];
    }
    __syncthreads();

    const uint4 zero4 = make_uint4(0u, 0u, 0u, 0u);

    for (int tap = 0; tap < 27; tap++) {
        const int buf = tap & 1;
        // prefetch next tap's B slice into the other LDS buffer
        if (tap + 1 < 27) {
            const uint4* src = (const uint4*)(wpack + (size_t)(tap + 1) * 8192);
            uint4* dst = (uint4*)(&lds[(tap + 1) & 1][0]);
            #pragma unroll
            for (int i = 0; i < 4; i++) dst[tid * 4 + i] = src[tid * 4 + i];
        }

        // A fragments: lane loads its 16 halves of xr and xi for this tap.
        const int dz = tap / 9 - 1, dy = (tap / 3) % 3 - 1, dx = tap % 3 - 1;
        const int zz = d0 + dz, yy = h0 + dy, xx = w0 + dx;
        const bool valid = ((unsigned)zz < 32u) & ((unsigned)yy < 32u) & ((unsigned)xx < 32u);
        const int zc = min(max(zz, 0), 31), yc = min(max(yy, 0), 31), xc = min(max(xx, 0), 31);

        const size_t rowBase = (((size_t)zc * 32 + yc) * 32 + xc) * 32 + kOff;
        const uint4* pr = (const uint4*)(xrh + rowBase);
        const uint4* pi = (const uint4*)(xih + rowBase);

        // K groups {kOff..kOff+7} and {kOff+16..kOff+23}: uint4 index 0 and 2
        union { v16h h; uint4 q[2]; } ar, ai;
        uint4 r0 = pr[0], r1 = pr[2];
        uint4 i0 = pi[0], i1 = pi[2];
        ar.q[0] = valid ? r0 : zero4;   // branchless zero-pad (v_cndmask),
        ar.q[1] = valid ? r1 : zero4;   // EXEC stays all-1s for WMMA
        ai.q[0] = valid ? i0 : zero4;
        ai.q[1] = valid ? i1 : zero4;

        // hint the next tap's activation line toward the caches
        __builtin_prefetch(xrh + rowBase + 1024, 0, 0);
        __builtin_prefetch(xih + rowBase + 1024, 0, 0);

        #pragma unroll
        for (int nt = 0; nt < 8; nt++) {
            v16h b0 = *(const v16h*)(&lds[buf][((0 * 8 + nt) * 32 + lane) * 16]);
            v16h b1 = *(const v16h*)(&lds[buf][((1 * 8 + nt) * 32 + lane) * 16]);
            acc[nt] = __builtin_amdgcn_wmma_f32_16x16x32_f16(
                false, ar.h, false, b0, (short)0, acc[nt], false, false);
            acc[nt] = __builtin_amdgcn_wmma_f32_16x16x32_f16(
                false, ai.h, false, b1, (short)0, acc[nt], false, false);
        }
        __syncthreads();   // next-tap staging done; safe to reuse buffers
    }

    // Store: D tile layout -> VGPR r holds (M=r, N=lane) for lanes 0-15 and
    // (M=r+8, N=lane-16) for lanes 16-31. Pair yr (nt) with yi (nt+4) as
    // interleaved complex float2, channel-last (pos*64 + o).
    #pragma unroll
    for (int nt = 0; nt < 4; nt++) {
        #pragma unroll
        for (int r = 0; r < 8; r++) {
            int pos = tileBase + r + hiHalf * 8;
            int o   = nt * 16 + laneM;
            float2 v;
            v.x = acc[nt][r];       // real part
            v.y = acc[nt + 4][r];   // imag part
            *(float2*)(out + ((size_t)pos * 64 + o) * 2) = v;
        }
    }
}

// ---------------------------------------------------------------------------
extern "C" void kernel_launch(void* const* d_in, const int* in_sizes, int n_in,
                              void* d_out, int out_size, void* d_ws, size_t ws_size,
                              hipStream_t stream)
{
    const float* xr   = (const float*)d_in[0];
    const float* xi   = (const float*)d_in[1];
    const float* w00r = (const float*)d_in[2],  *w00i = (const float*)d_in[3];
    const float* w01r = (const float*)d_in[4],  *w01i = (const float*)d_in[5];
    const float* w10r = (const float*)d_in[6],  *w10i = (const float*)d_in[7];
    const float* w11r = (const float*)d_in[8],  *w11i = (const float*)d_in[9];
    const float* f00r = (const float*)d_in[10], *f00i = (const float*)d_in[11];
    const float* f10r = (const float*)d_in[12], *f10i = (const float*)d_in[13];
    const float* f01r = (const float*)d_in[14], *f01i = (const float*)d_in[15];
    const float* f11r = (const float*)d_in[16], *f11i = (const float*)d_in[17];

    float*     out   = (float*)d_out;
    float*     kr    = (float*)d_ws;                    // 55296 f32
    float*     ki    = kr + 55296;                      // 55296 f32
    _Float16*  wpack = (_Float16*)(ki + 55296);         // 221184 f16
    _Float16*  xrh   = wpack + 221184;                  // 1048576 f16 (2 MB)
    _Float16*  xih   = xrh + 1048576;                   // 1048576 f16 (2 MB)

    se3_build_kernel<<<(55296 + 255) / 256, 256, 0, stream>>>(
        w00r, w00i, w01r, w01i, w10r, w10i, w11r, w11i,
        f00r, f00i, f10r, f10i, f01r, f01i, f11r, f11i, kr, ki);

    se3_pack_weights<<<(13824 + 255) / 256, 256, 0, stream>>>(kr, ki, wpack);

    se3_cvt_x_f16<<<262144 / 256, 256, 0, stream>>>(xr, xi, xrh, xih);

    // 32768 spatial rows / (8 waves * 16 rows) = 256 workgroups of 256 threads
    se3_conv_wmma<<<256, 256, 0, stream>>>(xrh, xih, wpack, out);
}
